// MessagePassingGNN_75376676045019
// MI455X (gfx1250) — matrix operand order repaired and checked by reading, not compiled, not run on hardware
//
#include <hip/hip_runtime.h>

// ---------------------------------------------------------------------------
// GAT forward on gfx1250 (MI455X), compile-only tuned.
//  - adj (256MB) bit-packed once to 8MB -> L2-resident for all softmax passes
//  - all GEMMs (proj, Wh, attn@Wh) via v_wmma_f32_16x16x32_bf16, f32 accum
//  - two-pass masked softmax (row max/sumexp stats, then normalized P@Wh)
//  - aggregation streams B tiles through LDS with the Tensor Data Mover
//    (double buffered, TENSORcnt-synchronized), LDS rows padded by TDM
//    (+4 dwords / 64B row) for bank-conflict-free ds_load_b128 fragments
// ---------------------------------------------------------------------------

#define N_NODES 8192
#define D_IN    256
#define H_DIM   128
#define O_DIM   64
#define ALPHA   0.2f
#define NEGV    (-9e15f)

typedef __attribute__((ext_vector_type(16))) __bf16 v16bf;
typedef __attribute__((ext_vector_type(8)))  float  v8f;
typedef __attribute__((ext_vector_type(4)))  unsigned u32x4;
typedef __attribute__((ext_vector_type(8)))  int    i32x8;
typedef __attribute__((ext_vector_type(4)))  int    i32x4;

#if defined(__AMDGCN__) && __has_builtin(__builtin_amdgcn_tensor_load_to_lds) && \
    __has_builtin(__builtin_amdgcn_s_wait_tensorcnt)
#define HAVE_TDM 1
#else
#define HAVE_TDM 0
#endif

// ---------------------------------------------------------------------------
// K0: pack adj[8192][8192] int32 -> bitmask[8192][256] u32 (bit j of word j/32)
// ---------------------------------------------------------------------------
__global__ void pack_adj_kernel(const int* __restrict__ adj,
                                unsigned* __restrict__ bits) {
    __shared__ unsigned w[8];
    const int t = threadIdx.x;                 // 256 threads
    if (t < 8) w[t] = 0u;
    __syncthreads();
    const size_t base = (size_t)blockIdx.x * 256u;
    if (adj[base + t] > 0) atomicOr(&w[t >> 5], 1u << (t & 31));
    __syncthreads();
    if (t < 8) bits[(size_t)blockIdx.x * 8u + t] = w[t];
}

// ---------------------------------------------------------------------------
// K1: h = relu(x @ Wp.T + bp) -> bf16 [N, H]. One wave -> 16x16 tile, K=256.
// A frag (lane L): row m0+(L&15); ks = (L>>4)*16 + {0..15}
// B frag (lane L): col n0+(L&15); ks = (L>>4)*16 + {0..15}
// ---------------------------------------------------------------------------
__global__ void proj_kernel(const float* __restrict__ x,
                            const float* __restrict__ Wp,
                            const float* __restrict__ bp,
                            __bf16* __restrict__ hbf) {
    const int lane = threadIdx.x & 31;
    const int wave = threadIdx.x >> 5;
    const int m0   = blockIdx.x * 16;
    const int n0   = blockIdx.y * 64 + wave * 16;
    const int half = lane >> 4;
    const int l16  = lane & 15;
    const int arow = m0 + l16;
    const int bcol = n0 + l16;

    v8f acc = {};
    for (int kk = 0; kk < D_IN; kk += 32) {
        v16bf a, b;
        const float* xp = x + arow * D_IN + kk + half * 16;
#pragma unroll
        for (int e = 0; e < 8; ++e) {
            a[e]     = (__bf16)xp[e];
            a[e + 8] = (__bf16)xp[e + 8];
        }
        const float* wp = Wp + bcol * D_IN + kk + half * 16;
#pragma unroll
        for (int e = 0; e < 16; ++e) b[e] = (__bf16)wp[e];
        acc = __builtin_amdgcn_wmma_f32_16x16x32_bf16(false, a, false, b,
                                                      (short)0, acc, false, false);
    }
    const float bias = bp[bcol];
#pragma unroll
    for (int v = 0; v < 8; ++v) {
        const int m = v + half * 8;              // D layout: M = v + 8*(lane/16)
        float r = acc[v] + bias;
        r = r > 0.f ? r : 0.f;
        hbf[(m0 + m) * H_DIM + n0 + l16] = (__bf16)r;
    }
}

// ---------------------------------------------------------------------------
// K2: WhT = (Hbf @ W.T)^T stored column-major: WhT[col][row] bf16 [F][N].
// Transposed store makes aggregation B-fragments contiguous along K.
// ---------------------------------------------------------------------------
__global__ void gemm_whT_kernel(const __bf16* __restrict__ hbf,
                                const float* __restrict__ W,
                                __bf16* __restrict__ whT) {
    const int lane = threadIdx.x & 31;
    const int wave = threadIdx.x >> 5;
    const int m0   = blockIdx.x * 16;
    const int n0   = blockIdx.y * 64 + wave * 16;
    const int half = lane >> 4;
    const int l16  = lane & 15;
    const int arow = m0 + l16;
    const int bcol = n0 + l16;

    v8f acc = {};
    for (int kk = 0; kk < H_DIM; kk += 32) {
        v16bf a, b;
        const __bf16* hp = hbf + arow * H_DIM + kk + half * 16;
#pragma unroll
        for (int e = 0; e < 8; ++e) {
            a[e]     = hp[e];
            a[e + 8] = hp[e + 8];
        }
        const float* wp = W + bcol * H_DIM + kk + half * 16;
#pragma unroll
        for (int e = 0; e < 16; ++e) b[e] = (__bf16)wp[e];
        acc = __builtin_amdgcn_wmma_f32_16x16x32_bf16(false, a, false, b,
                                                      (short)0, acc, false, false);
    }
#pragma unroll
    for (int v = 0; v < 8; ++v) {
        const int m = v + half * 8;
        whT[(size_t)(n0 + l16) * N_NODES + m0 + m] = (__bf16)acc[v];
    }
}

// ---------------------------------------------------------------------------
// K3: s_src[i] = Wh[i,:].a[:F] ; s_dst[i] = Wh[i,:].a[F:]  (coalesced over i)
// ---------------------------------------------------------------------------
__global__ void scores_kernel(const __bf16* __restrict__ whT,
                              const float* __restrict__ a,
                              float* __restrict__ s_src,
                              float* __restrict__ s_dst, int F) {
    const int i = blockIdx.x * blockDim.x + threadIdx.x;
    float as = 0.f, ad = 0.f;
    for (int c = 0; c < F; ++c) {
        const float w = (float)whT[(size_t)c * N_NODES + i];
        as += w * a[c];
        ad += w * a[F + c];
    }
    s_src[i] = as;
    s_dst[i] = ad;
}

// cross-lane xor shuffle via ds_bpermute (avoid header wavefront-size deps)
__device__ inline float xor_bpermute(float v, int lane, int mask) {
    const int idx = ((lane ^ mask) << 2);
    return __int_as_float(__builtin_amdgcn_ds_bpermute(idx, __float_as_int(v)));
}

// ---------------------------------------------------------------------------
// K4: per-row softmax stats. 4 rows per wave: one s_dst load feeds 4 rows,
// cutting the dominant L2 read stream 4x. Masked entries use NEGV (matches
// reference jnp.where semantics, incl. fully-masked rows -> uniform softmax).
// ---------------------------------------------------------------------------
__global__ void stats_kernel(const unsigned* __restrict__ bits,
                             const float* __restrict__ s_src,
                             const float* __restrict__ s_dst,
                             float* __restrict__ mrow,
                             float* __restrict__ lrow) {
    const int lane = threadIdx.x & 31;
    const int i0   = (blockIdx.x * 4 + (threadIdx.x >> 5)) * 4;

    float ss[4];
    const unsigned* br[4];
#pragma unroll
    for (int q = 0; q < 4; ++q) {
        ss[q] = s_src[i0 + q];
        br[q] = bits + (size_t)(i0 + q) * 256u;
    }

    float m[4] = {-3.4e38f, -3.4e38f, -3.4e38f, -3.4e38f};
    for (int jb = 0; jb < N_NODES; jb += 32) {
        const float sd = s_dst[jb + lane];
#pragma unroll
        for (int q = 0; q < 4; ++q) {
            const unsigned w = br[q][jb >> 5];
            float e = ss[q] + sd;
            e = e > 0.f ? e : ALPHA * e;
            e = ((w >> lane) & 1u) ? e : NEGV;
            m[q] = fmaxf(m[q], e);
        }
    }
#pragma unroll
    for (int q = 0; q < 4; ++q)
#pragma unroll
        for (int off = 16; off > 0; off >>= 1)
            m[q] = fmaxf(m[q], xor_bpermute(m[q], lane, off));

    float l[4] = {0.f, 0.f, 0.f, 0.f};
    for (int jb = 0; jb < N_NODES; jb += 32) {
        const float sd = s_dst[jb + lane];
#pragma unroll
        for (int q = 0; q < 4; ++q) {
            const unsigned w = br[q][jb >> 5];
            float e = ss[q] + sd;
            e = e > 0.f ? e : ALPHA * e;
            e = ((w >> lane) & 1u) ? e : NEGV;
            l[q] += __expf(e - m[q]);
        }
    }
#pragma unroll
    for (int q = 0; q < 4; ++q)
#pragma unroll
        for (int off = 16; off > 0; off >>= 1)
            l[q] += xor_bpermute(l[q], lane, off);

    if (lane == 0) {
#pragma unroll
        for (int q = 0; q < 4; ++q) { mrow[i0 + q] = m[q]; lrow[i0 + q] = l[q]; }
    }
}

#if HAVE_TDM
// Issue a 2D TDM load: tile = WhT[0..F_rows-1][k0..k0+31] (bf16, row stride
// N_NODES) into LDS at lds_byte_off, with +4dword pad per 16-dword (64B) row
// => padded LDS row stride of 80B (bank-conflict-free, 16B aligned).
__device__ __forceinline__ void tdm_issue_whT_tile(unsigned lds_byte_off,
                                                   const __bf16* gptr,
                                                   int F_rows) {
    const unsigned long long ga = (unsigned long long)gptr;
    u32x4 g0;
    g0[0] = 1u;                                            // count=1, user mode
    g0[1] = lds_byte_off;                                  // lds_addr
    g0[2] = (unsigned)ga;                                  // global_addr[31:0]
    g0[3] = (unsigned)((ga >> 32) & 0x01FFFFFFu) | (2u << 30); // addr[56:32]|type=2
    i32x8 g1;
    g1[0] = (1 << 16) | (1 << 20) | (3 << 22) | (3 << 25); // 2B elems, pad_en,
                                                           // interval=16dw, amt=4dw
    g1[1] = (int)(0x2000u << 16);                          // tensor_dim0=8192 lo16
    g1[2] = (int)((unsigned)F_rows << 16);                 // dim0 hi16=0 | dim1 lo16
    g1[3] = (32 << 16);                                    // dim1 hi16=0 | tile_dim0=32
    g1[4] = F_rows;                                        // tile_dim1 | tile_dim2=0
    g1[5] = (int)N_NODES;                                  // tensor_dim0_stride lo32
    g1[6] = 0;
    g1[7] = 0;
    i32x4 z4 = {};
#if __clang_major__ >= 23
    i32x8 z8 = {};
    __builtin_amdgcn_tensor_load_to_lds(g0, g1, z4, z4, z8, 0);
#else
    __builtin_amdgcn_tensor_load_to_lds(g0, g1, z4, z4, 0);
#endif
}
#endif

// ---------------------------------------------------------------------------
// K5: h' = softmax(e) @ Wh via WMMA. Block = 128 threads (4 waves), 32 rows,
// F = NW*64 cols (wave w -> NW 16-col fragments x 2 row-groups).
// Per 32-wide j-tile: P(32x32) computed in f32 (exp), staged bf16 in padded
// LDS; B tile (F x 32) TDM-streamed into double-buffered padded LDS.
// FINAL=false: ELU + bf16 store (next layer input). FINAL=true: f32 -> d_out.
// ---------------------------------------------------------------------------
template <int NW, bool FINAL>
__global__ void __launch_bounds__(128)
aggregate_kernel(const unsigned* __restrict__ bits,
                 const float* __restrict__ s_src,
                 const float* __restrict__ s_dst,
                 const float* __restrict__ mrow,
                 const float* __restrict__ lrow,
                 const __bf16* __restrict__ whT,
                 void* __restrict__ outp) {
    constexpr int F    = NW * 64;
    constexpr int BROW = 40;                     // padded row stride in elements
    __shared__ __bf16 pT[32 * BROW];             // P tile, padded rows
#if HAVE_TDM
    __shared__ __bf16 bbuf[2][F * BROW];         // double-buffered B tiles
#endif

    const int lane = threadIdx.x & 31;
    const int wave = threadIdx.x >> 5;
    const int i0   = blockIdx.x * 32;

    // staging mapping: thread handles k = tid&31, rows rq + 4*q (q=0..7)
    const int k  = threadIdx.x & 31;
    const int rq = threadIdx.x >> 5;
    float ssr[8], mr8[8];
    const unsigned* brp[8];
#pragma unroll
    for (int q = 0; q < 8; ++q) {
        const int r = rq + 4 * q;
        ssr[q] = s_src[i0 + r];
        mr8[q] = mrow[i0 + r];
        brp[q] = bits + (size_t)(i0 + r) * 256u;
    }

    const int half = lane >> 4;
    const int l16  = lane & 15;
    const int c0   = wave * (NW * 16);

    v8f zero = {};
    v8f acc[2][NW];
#pragma unroll
    for (int g = 0; g < 2; ++g)
#pragma unroll
        for (int c = 0; c < NW; ++c) acc[g][c] = zero;

#if HAVE_TDM
    unsigned bofs[2];
    bofs[0] = (unsigned)(size_t)&bbuf[0][0];     // flat low 32 bits == LDS offset
    bofs[1] = (unsigned)(size_t)&bbuf[1][0];
    if (wave == 0) tdm_issue_whT_tile(bofs[0], whT, F);   // prologue: tile 0
#endif

    int tile = 0;
    for (int jb = 0; jb < N_NODES; jb += 32, ++tile) {
#if HAVE_TDM
        const bool more = (jb + 32) < N_NODES;
        if (wave == 0 && more)                    // prefetch next tile
            tdm_issue_whT_tile(bofs[(tile + 1) & 1], whT + (jb + 32), F);
#endif
        // ---- stage P tile (32x32), exp in f32, bf16 into padded LDS ----
        const float sd = s_dst[jb + k];
#pragma unroll
        for (int q = 0; q < 8; ++q) {
            const int r = rq + 4 * q;
            const unsigned w = brp[q][jb >> 5];
            float e = ssr[q] + sd;
            e = e > 0.f ? e : ALPHA * e;
            e = ((w >> k) & 1u) ? e : NEGV;
            pT[r * BROW + k] = (__bf16)__expf(e - mr8[q]);  // /l in epilogue
        }
#if HAVE_TDM
        if (more) __builtin_amdgcn_s_wait_tensorcnt(1);     // tile `tile` done
        else      __builtin_amdgcn_s_wait_tensorcnt(0);
        const __bf16* bb = &bbuf[tile & 1][0];
#endif
        __syncthreads();                          // P + B tile visible

#pragma unroll
        for (int g = 0; g < 2; ++g) {
            v16bf a;
            const __bf16* ap = &pT[(g * 16 + l16) * BROW + half * 16];
#pragma unroll
            for (int e = 0; e < 8; ++e) {
                a[e]     = ap[e];
                a[e + 8] = ap[e + 8];
            }
#pragma unroll
            for (int c = 0; c < NW; ++c) {
#if HAVE_TDM
                const __bf16* bp = &bb[(c0 + c * 16 + l16) * BROW + half * 16];
#else
                const __bf16* bp = whT + (size_t)(c0 + c * 16 + l16) * N_NODES
                                       + jb + half * 16;
#endif
                v16bf b;
#pragma unroll
                for (int e = 0; e < 16; ++e) b[e] = bp[e];
                acc[g][c] = __builtin_amdgcn_wmma_f32_16x16x32_bf16(
                    false, a, false, b, (short)0, acc[g][c], false, false);
            }
        }
        __syncthreads();                          // pT / bbuf reuse safe
    }

    // ---- epilogue: normalize by l, (ELU+bf16) or f32 out ----
#pragma unroll
    for (int g = 0; g < 2; ++g)
#pragma unroll
        for (int v = 0; v < 8; ++v) {
            const int m = g * 16 + v + half * 8;  // D layout M within 32 rows
            const float invl = 1.0f / lrow[i0 + m];
#pragma unroll
            for (int c = 0; c < NW; ++c) {
                float r = acc[g][c][v] * invl;
                const int col = c0 + c * 16 + l16;
                if (FINAL) {
                    ((float*)outp)[(size_t)(i0 + m) * F + col] = r;
                } else {
                    r = r > 0.f ? r : (__expf(r) - 1.0f);   // ELU
                    ((__bf16*)outp)[(size_t)(i0 + m) * F + col] = (__bf16)r;
                }
            }
        }
}

// ---------------------------------------------------------------------------
extern "C" void kernel_launch(void* const* d_in, const int* in_sizes, int n_in,
                              void* d_out, int out_size, void* d_ws, size_t ws_size,
                              hipStream_t stream) {
    const float* x   = (const float*)d_in[0];   // [8192,256]
    const int*   adj = (const int*)  d_in[1];   // [8192,8192]
    const float* Wp  = (const float*)d_in[2];   // [128,256]
    const float* bp  = (const float*)d_in[3];   // [128]
    const float* W1  = (const float*)d_in[4];   // [128,128]
    const float* a1  = (const float*)d_in[5];   // [256]
    const float* W2  = (const float*)d_in[6];   // [64,128]
    const float* a2  = (const float*)d_in[7];   // [128]

    char* ws = (char*)d_ws;
    size_t off = 0;
    unsigned* bits = (unsigned*)(ws + off); off += (size_t)N_NODES * 256 * 4;  // 8 MB
    __bf16* h1     = (__bf16*) (ws + off);  off += (size_t)N_NODES * H_DIM * 2;
    __bf16* whT1   = (__bf16*) (ws + off);  off += (size_t)H_DIM * N_NODES * 2;
    __bf16* h2     = (__bf16*) (ws + off);  off += (size_t)N_NODES * H_DIM * 2;
    __bf16* whT2   = (__bf16*) (ws + off);  off += (size_t)O_DIM * N_NODES * 2;
    float* ssrc1 = (float*)(ws + off); off += N_NODES * 4;
    float* sdst1 = (float*)(ws + off); off += N_NODES * 4;
    float* m1    = (float*)(ws + off); off += N_NODES * 4;
    float* l1    = (float*)(ws + off); off += N_NODES * 4;
    float* ssrc2 = (float*)(ws + off); off += N_NODES * 4;
    float* sdst2 = (float*)(ws + off); off += N_NODES * 4;
    float* m2    = (float*)(ws + off); off += N_NODES * 4;
    float* l2    = (float*)(ws + off); off += N_NODES * 4;

    // K0: bit-pack adjacency (one-time 256MB read; mask lives in L2 after)
    pack_adj_kernel<<<(N_NODES * (size_t)N_NODES) / 256, 256, 0, stream>>>(adj, bits);

    // K1: input projection + relu -> bf16
    proj_kernel<<<dim3(N_NODES / 16, H_DIM / 64), 128, 0, stream>>>(x, Wp, bp, h1);

    // ---- layer 1 (F = 128, ELU) ----
    gemm_whT_kernel<<<dim3(N_NODES / 16, H_DIM / 64), 128, 0, stream>>>(h1, W1, whT1);
    scores_kernel<<<N_NODES / 256, 256, 0, stream>>>(whT1, a1, ssrc1, sdst1, H_DIM);
    stats_kernel<<<N_NODES / 16, 128, 0, stream>>>(bits, ssrc1, sdst1, m1, l1);
    aggregate_kernel<2, false><<<N_NODES / 32, 128, 0, stream>>>(
        bits, ssrc1, sdst1, m1, l1, whT1, h2);

    // ---- layer 2 (F = 64, no ELU, f32 output) ----
    gemm_whT_kernel<<<dim3(N_NODES / 16, O_DIM / 64), 128, 0, stream>>>(h2, W2, whT2);
    scores_kernel<<<N_NODES / 256, 256, 0, stream>>>(whT2, a2, ssrc2, sdst2, O_DIM);
    stats_kernel<<<N_NODES / 16, 128, 0, stream>>>(bits, ssrc2, sdst2, m2, l2);
    aggregate_kernel<1, true><<<N_NODES / 32, 128, 0, stream>>>(
        bits, ssrc2, sdst2, m2, l2, whT2, d_out);
}
